// total_communicability_pool_60902636257964
// MI455X (gfx1250) — matrix operand order repaired and checked by reading, not compiled, not run on hardware
//
#include <hip/hip_runtime.h>
#include <cstdint>

// ---------------------------------------------------------------------------
// MI455X (gfx1250) implementation of total-communicability top-k pooling.
// Roofline: ~350 MB of traffic per call, ~15-25us at 23.3 TB/s; every stage is
// bandwidth bound. The GEMV (x@W) is mapped onto V_WMMA_F32_16X16X32_F16 so
// the score stage runs on the CDNA5 matrix pipe while streaming x at full BW.
// B fragments (W broadcast into column 0) are prebuilt in VGPRs so the hot
// loop is pure: 4x global_load_b128 -> 8x v_cvt_pk_f16_f32 -> v_wmma.
// ---------------------------------------------------------------------------

typedef __attribute__((ext_vector_type(16))) _Float16 v16h;
typedef __attribute__((ext_vector_type(8)))  float    v8f;

#define D_FEAT  256
#define NBUCKET 2048

__device__ __forceinline__ unsigned order_key(float f) {
  // monotonic float->uint transform (larger key <=> larger float)
  unsigned u = __float_as_uint(f);
  return (u & 0x80000000u) ? ~u : (u | 0x80000000u);
}

// ---- 1) score = x @ W + b, one wave handles a 16-row tile via WMMA --------
__global__ void score_wmma_kernel(const float* __restrict__ x,
                                  const float* __restrict__ W,
                                  const float* __restrict__ bias,
                                  float* __restrict__ tc,
                                  float* __restrict__ cur,
                                  int nrows) {
  const int lane    = threadIdx.x & 31;
  const int wave    = (int)((blockIdx.x * blockDim.x + threadIdx.x) >> 5);
  const int rowBase = wave << 4;
  if (rowBase >= nrows) return;                 // uniform per wave (EXEC stays full)

  const int m   = lane & 15;                    // row within tile (A) / column (B,D)
  const int hi  = lane >> 4;                    // lane half select
  const int row = rowBase + m;
  const int kA0 = hi << 3;                      // A: lanes 0-15 -> K {0..7,16..23}
                                                //    lanes 16-31 -> K {8..15,24..31}

  // Prebuild all 8 B fragments (32x16 each): column 0 = W[kb..kb+31], rest 0.
  // Branchless masking keeps EXEC untouched near the WMMAs.
  const float cm = (m == 0) ? 1.0f : 0.0f;
  v16h bfr[8];
#pragma unroll
  for (int t = 0; t < 8; ++t) {
    const float* wp = W + t * 32 + (hi << 3);   // lanes 0-15: K {0..7,16..23}, 16-31: K {8..15,24..31}
#pragma unroll
    for (int h = 0; h < 16; ++h) bfr[t][h] = (_Float16)(cm * wp[h < 8 ? h : h + 8]);
  }

  v8f acc = {};
  const float* xrow = x + (size_t)row * D_FEAT;
#pragma unroll
  for (int t = 0; t < 8; ++t) {
    const float* xr = xrow + t * 32;
    v16h a;
#pragma unroll
    for (int h = 0; h < 8; ++h) a[h]     = (_Float16)xr[kA0 + h];
#pragma unroll
    for (int h = 0; h < 8; ++h) a[8 + h] = (_Float16)xr[kA0 + 16 + h];
    // 8-arg confirmed form: (neg_a, A, neg_b, B, c_mod, C, reuse_a, reuse_b)
    acc = __builtin_amdgcn_wmma_f32_16x16x32_f16(false, a, false, bfr[t],
                                                 (short)0, acc, false, false);
  }

  // D column 0: lane 0 holds M=0..7 in acc[0..7], lane 16 holds M=8..15.
  if (m == 0) {
    const float b0 = bias[0];
#pragma unroll
    for (int r = 0; r < 8; ++r) {
      int rr = rowBase + (hi << 3) + r;
      if (rr < nrows) {
        float s = acc[r] + b0;
        tc[rr]  = s;      // running total-communicability accumulator
        cur[rr] = s;      // current hop vector
      }
    }
  }
}

// ---- 2) K-hop propagation -------------------------------------------------
__global__ void zero_f32_kernel(float* __restrict__ p, int n) {
  int i = blockIdx.x * blockDim.x + threadIdx.x;
  if (i < n) p[i] = 0.0f;
}

__global__ void zero_u32_kernel(unsigned* __restrict__ p, int n) {
  int i = blockIdx.x * blockDim.x + threadIdx.x;
  if (i < n) p[i] = 0u;
}

__global__ void edge_scatter_kernel(const long long* __restrict__ src,
                                    const long long* __restrict__ dst,
                                    const float* __restrict__ cur,
                                    float* __restrict__ nxt, int E) {
  int e = blockIdx.x * blockDim.x + threadIdx.x;
  if (e < E) {
    int s = (int)src[e];
    int d = (int)dst[e];
    atomicAdd(&nxt[d], cur[s]);
  }
}

__global__ void accum_tc_kernel(float* __restrict__ tc,
                                const float* __restrict__ nxt,
                                float invfact, int n) {
  int i = blockIdx.x * blockDim.x + threadIdx.x;
  if (i < n) tc[i] += nxt[i] * invfact;
}

// ---- 3) top-k via radix histogram threshold -------------------------------
__global__ void hist_kernel(const float* __restrict__ tc,
                            unsigned* __restrict__ hist, int n) {
  __shared__ unsigned lh[NBUCKET];
  for (int i = threadIdx.x; i < NBUCKET; i += blockDim.x) lh[i] = 0u;
  __syncthreads();
  int i      = blockIdx.x * blockDim.x + threadIdx.x;
  int stride = gridDim.x * blockDim.x;
  for (; i < n; i += stride) atomicAdd(&lh[order_key(tc[i]) >> 21], 1u);
  __syncthreads();
  for (int i = threadIdx.x; i < NBUCKET; i += blockDim.x)
    if (lh[i]) atomicAdd(&hist[i], lh[i]);
}

__global__ void find_thresh_kernel(const unsigned* __restrict__ hist,
                                   int* __restrict__ params, int kkeep) {
  if (threadIdx.x == 0 && blockIdx.x == 0) {
    unsigned cum = 0;
    for (int b = NBUCKET - 1; b >= 0; --b) {
      unsigned c = hist[b];
      if (cum + c >= (unsigned)kkeep) {
        params[0] = b;                      // threshold bucket
        params[1] = kkeep - (int)cum;       // quota to take from bucket b
        return;
      }
      cum += c;
    }
    params[0] = 0;
    params[1] = kkeep - (int)cum;
  }
}

__global__ void select_kernel(const float* __restrict__ tc,
                              int* __restrict__ newidx,
                              int* __restrict__ perm,
                              int* __restrict__ counters,   // [0]=kept, [1]=ties seen
                              const int* __restrict__ params,
                              int n) {
  int i = blockIdx.x * blockDim.x + threadIdx.x;
  if (i >= n) return;
  int bkt = (int)(order_key(tc[i]) >> 21);
  int T = params[0], quota = params[1];
  bool keep = false;
  if (bkt > T) keep = true;
  else if (bkt == T) {
    int t = atomicAdd(&counters[1], 1);
    keep = (t < quota);
  }
  if (keep) {
    int p = atomicAdd(&counters[0], 1);
    perm[p]   = i;
    newidx[i] = p;
  } else {
    newidx[i] = -1;
  }
}

// ---- 4) gather kept rows scaled by tanh(score) ----------------------------
__global__ void gather_scale_kernel(const float* __restrict__ x,
                                    const int* __restrict__ perm,
                                    const float* __restrict__ tc,
                                    float* __restrict__ x_out) {
  int p = blockIdx.x;
  int d = threadIdx.x;                 // blockDim == D_FEAT
  int row = perm[p];
  float s = tanhf(tc[row]);
  x_out[(size_t)p * D_FEAT + d] = x[(size_t)row * D_FEAT + d] * s;
}

// ---- 5) filter + remap edges ----------------------------------------------
__global__ void remap_edges_kernel(const long long* __restrict__ src,
                                   const long long* __restrict__ dst,
                                   const int* __restrict__ newidx,
                                   long long* __restrict__ out_src,
                                   long long* __restrict__ out_dst, int E) {
  int e = blockIdx.x * blockDim.x + threadIdx.x;
  if (e >= E) return;
  int s = (int)src[e], d = (int)dst[e];
  int ns = newidx[s], nd = newidx[d];
  bool valid = (ns >= 0) && (nd >= 0);
  out_src[e] = valid ? (long long)ns : -1ll;
  out_dst[e] = valid ? (long long)nd : -1ll;
}

// ---- 6) batch zeros + perm output -----------------------------------------
__global__ void write_batch_perm_kernel(long long* __restrict__ batch,
                                        int* __restrict__ perm_out,
                                        const int* __restrict__ perm, int kkeep) {
  int i = blockIdx.x * blockDim.x + threadIdx.x;
  if (i < kkeep) {
    batch[i]    = 0ll;
    perm_out[i] = perm[i];
  }
}

// ---------------------------------------------------------------------------
extern "C" void kernel_launch(void* const* d_in, const int* in_sizes, int n_in,
                              void* d_out, int out_size, void* d_ws, size_t ws_size,
                              hipStream_t stream) {
  const float*     x  = (const float*)d_in[0];
  const long long* ei = (const long long*)d_in[1];
  const float*     W  = (const float*)d_in[2];
  const float*     b  = (const float*)d_in[3];

  const int N     = in_sizes[0] / D_FEAT;
  const int E     = in_sizes[1] / 2;
  const int KKEEP = (N + 1) / 2;
  const long long* src = ei;
  const long long* dst = ei + E;

  // workspace carve-out (all state re-initialized every call)
  char*  ws  = (char*)d_ws;
  size_t off = 0;
  auto alloc = [&](size_t bytes) -> void* {
    void* p = ws + off;
    off = (off + bytes + 255) & ~(size_t)255;
    return p;
  };
  float*    tc       = (float*)alloc((size_t)N * 4);
  float*    bufA     = (float*)alloc((size_t)N * 4);
  float*    bufB     = (float*)alloc((size_t)N * 4);
  int*      nidx     = (int*)alloc((size_t)N * 4);
  int*      perm     = (int*)alloc((size_t)KKEEP * 4);
  unsigned* hist     = (unsigned*)alloc((size_t)NBUCKET * 4);
  int*      params   = (int*)alloc(2 * 4);
  int*      counters = (int*)alloc(2 * 4);
  (void)ws_size; (void)n_in;

  // output layout: x_out f32 | edge_index_new i64 (2,E) | batch i64 | perm i32
  char*      outb      = (char*)d_out;
  float*     x_out     = (float*)outb;
  long long* e_out     = (long long*)(outb + (size_t)KKEEP * D_FEAT * 4);
  long long* batch_out = (long long*)(outb + (size_t)KKEEP * D_FEAT * 4 + (size_t)2 * E * 8);
  int*       perm_out  = (int*)(outb + (size_t)KKEEP * D_FEAT * 4 + (size_t)2 * E * 8 + (size_t)KKEEP * 8);
  (void)out_size;

  const int TB = 256;
  const int nb = (N + TB - 1) / TB;
  const int eb = (E + TB - 1) / TB;

  // 1) scores on the WMMA pipe
  {
    int waves  = (N + 15) / 16;
    int wpb    = 4;                                   // 128 threads/block
    int blocks = (waves + wpb - 1) / wpb;
    score_wmma_kernel<<<blocks, wpb * 32, 0, stream>>>(x, W, b, tc, bufA, N);
  }

  // 2) K=3 hops of A^k s / k!
  const float invfact[3] = {1.0f, 0.5f, 1.0f / 6.0f};
  float* cur = bufA;
  float* nxt = bufB;
  for (int hop = 0; hop < 3; ++hop) {
    zero_f32_kernel<<<nb, TB, 0, stream>>>(nxt, N);
    edge_scatter_kernel<<<eb, TB, 0, stream>>>(src, dst, cur, nxt, E);
    accum_tc_kernel<<<nb, TB, 0, stream>>>(tc, nxt, invfact[hop], N);
    float* t = cur; cur = nxt; nxt = t;
  }

  // 3) top-k threshold selection
  zero_u32_kernel<<<(NBUCKET + TB - 1) / TB, TB, 0, stream>>>(hist, NBUCKET);
  zero_u32_kernel<<<1, 32, 0, stream>>>((unsigned*)counters, 2);
  hist_kernel<<<512, TB, 0, stream>>>(tc, hist, N);
  find_thresh_kernel<<<1, 1, 0, stream>>>(hist, params, KKEEP);
  select_kernel<<<nb, TB, 0, stream>>>(tc, nidx, perm, counters, params, N);

  // 4) gather + tanh scale
  gather_scale_kernel<<<KKEEP, D_FEAT, 0, stream>>>(x, perm, tc, x_out);

  // 5) edge filter/remap
  remap_edges_kernel<<<eb, TB, 0, stream>>>(src, dst, nidx, e_out, e_out + E, E);

  // 6) batch + perm outputs
  write_batch_perm_kernel<<<(KKEEP + TB - 1) / TB, TB, 0, stream>>>(batch_out, perm_out, perm, KKEEP);
}
